// SpatialProcessor_7619271983412
// MI455X (gfx1250) — matrix operand order repaired and checked by reading, not compiled, not run on hardware
//
#include <hip/hip_runtime.h>
#include <hip/hip_bf16.h>
#include <math.h>

// Problem constants (match reference)
#define Bn 4
#define Nn 2048
#define IN_DIM 64
#define HID 128
#define ODIM 128
#define HEADS 4
#define DH 32           // head dim (HID/HEADS == ODIM/HEADS)
#define MW (Nn / 32)    // mask words per row (64)

typedef __attribute__((ext_vector_type(2)))  float  v2f;
typedef __attribute__((ext_vector_type(8)))  float  v8f;
typedef __attribute__((ext_vector_type(16))) __bf16 v16bf;

static __device__ __forceinline__ float wave_sum(float v) {
    #pragma unroll
    for (int o = 16; o > 0; o >>= 1) v += __shfl_xor(v, o, 32);
    return v;
}
static __device__ __forceinline__ float wave_max(float v) {
    #pragma unroll
    for (int o = 16; o > 0; o >>= 1) v = fmaxf(v, __shfl_xor(v, o, 32));
    return v;
}
static __device__ __forceinline__ float lrelu(float x) { return x > 0.0f ? x : 0.2f * x; }
static __device__ __forceinline__ float gelu_exact(float x) {
    return 0.5f * x * (1.0f + erff(x * 0.70710678118654752f));
}
static __device__ __forceinline__ unsigned short bf16_bits(float x) {
    return __builtin_bit_cast(unsigned short, (__bf16)x);
}

// ---------------------------------------------------------------------------
// 1) Row-normalize embedding (l2_normalize with eps=1e-12): ne[N][128] f32
// ---------------------------------------------------------------------------
__global__ void k_normalize(const float* __restrict__ emb, float* __restrict__ ne) {
    const int lane = threadIdx.x & 31;
    const int row  = (blockIdx.x * blockDim.x + threadIdx.x) >> 5;
    if (row >= Nn) return;
    const float* e = emb + row * HID;
    float v0 = e[lane], v1 = e[lane + 32], v2 = e[lane + 64], v3 = e[lane + 96];
    float ss = wave_sum(v0 * v0 + v1 * v1 + v2 * v2 + v3 * v3);
    float inv = rsqrtf(ss + 1e-12f);
    float* o = ne + row * HID;
    o[lane] = v0 * inv; o[lane + 32] = v1 * inv; o[lane + 64] = v2 * inv; o[lane + 96] = v3 * inv;
}

// ---------------------------------------------------------------------------
// 2) Adjacency mask via f32 WMMA (exact f32 -> threshold 0.5 -> bit pack).
//    One wave computes a 16(i) x 32(j) tile: two 16x16 accumulators, K=128.
// ---------------------------------------------------------------------------
__global__ void k_mask(const float* __restrict__ ne, unsigned* __restrict__ mask) {
    const int lane = threadIdx.x;
    const int ln = lane & 15, hl = lane >> 4;
    const int jt = blockIdx.x;   // 0..63  (32 cols each)
    const int it = blockIdx.y;   // 0..127 (16 rows each)
    const float* ri  = ne + (it * 16 + ln) * HID;
    const float* rjl = ne + (jt * 32 + ln) * HID;
    const float* rjr = ne + (jt * 32 + 16 + ln) * HID;
    v8f cl = {}, cr = {};
    #pragma unroll 4
    for (int k = 0; k < HID; k += 4) {
        const int ko = k + hl * 2;
        v2f a  = { ri[ko],  ri[ko + 1]  };
        v2f bl = { rjl[ko], rjl[ko + 1] };
        v2f br = { rjr[ko], rjr[ko + 1] };
        cl = __builtin_amdgcn_wmma_f32_16x16x4_f32(false, a, false, bl, (short)0, cl, false, false);
        cr = __builtin_amdgcn_wmma_f32_16x16x4_f32(false, a, false, br, (short)0, cr, false, false);
    }
    // C layout: reg r -> (M = r + 8*hl, Ncol = ln). Ballot packs 16 i-rows x 32 j-cols.
    #pragma unroll
    for (int r = 0; r < 8; ++r) {
        unsigned bL = (unsigned)__ballot(cl[r] > 0.5f);
        unsigned bR = (unsigned)__ballot(cr[r] > 0.5f);
        if (lane == 0) {
            mask[(it * 16 + r)     * MW + jt] = (bL & 0xFFFFu) | ((bR & 0xFFFFu) << 16);
            mask[(it * 16 + 8 + r) * MW + jt] = (bL >> 16)     | ((bR >> 16)     << 16);
        }
    }
}

// ---------------------------------------------------------------------------
// 3) Projection h = x @ Wp + bp  (f32 WMMA 16x16x4, K=64). rows = B*N.
// ---------------------------------------------------------------------------
__global__ void k_project(const float* __restrict__ x, const float* __restrict__ Wp,
                          const float* __restrict__ bp, float* __restrict__ h) {
    const int lane = threadIdx.x;
    const int ln = lane & 15, hl = lane >> 4;
    const int ft = blockIdx.x;   // 0..7
    const int rt = blockIdx.y;   // 0..511
    const int f = ft * 16 + ln;
    const float* ar = x + (rt * 16 + ln) * IN_DIM;
    v8f c = {};
    #pragma unroll 4
    for (int k = 0; k < IN_DIM; k += 4) {
        const int ko = k + hl * 2;
        v2f a = { ar[ko], ar[ko + 1] };
        v2f b = { Wp[ko * HID + f], Wp[(ko + 1) * HID + f] };
        c = __builtin_amdgcn_wmma_f32_16x16x4_f32(false, a, false, b, (short)0, c, false, false);
    }
    const float bias = bp[f];
    #pragma unroll
    for (int r = 0; r < 8; ++r) {
        const int m = r + 8 * hl;
        h[(rt * 16 + m) * HID + f] = c[r] + bias;
    }
}

// ---------------------------------------------------------------------------
// 4) Feature GEMM hh = h @ W (128x128, f32 WMMA), writes hh f32 and a per-head
//    transposed bf16 copy hhT[b][head][d][n] for WMMA B-fragment streaming.
// ---------------------------------------------------------------------------
__global__ void k_hidden(const float* __restrict__ hin, const float* __restrict__ W,
                         float* __restrict__ hh, unsigned short* __restrict__ hhT) {
    const int lane = threadIdx.x;
    const int ln = lane & 15, hl = lane >> 4;
    const int ft = blockIdx.x;   // 0..7
    const int nt = blockIdx.y;   // 0..127 (nodes within batch)
    const int b  = blockIdx.z;   // 0..3
    const int f = ft * 16 + ln;
    const float* ar = hin + ((size_t)(b * Nn + nt * 16 + ln)) * HID;
    v8f c = {};
    #pragma unroll 4
    for (int k = 0; k < HID; k += 4) {
        const int ko = k + hl * 2;
        v2f a = { ar[ko], ar[ko + 1] };
        v2f bb = { W[ko * HID + f], W[(ko + 1) * HID + f] };
        c = __builtin_amdgcn_wmma_f32_16x16x4_f32(false, a, false, bb, (short)0, c, false, false);
    }
    unsigned short pk[8];
    #pragma unroll
    for (int r = 0; r < 8; ++r) {
        const int m = r + 8 * hl;
        hh[((size_t)(b * Nn + nt * 16 + m)) * HID + f] = c[r];
        pk[r] = bf16_bits(c[r]);
    }
    // hhT: head = f>>5, d = f&31; 8 consecutive n's per lane -> one b128 store
    const int head = f >> 5, d = f & 31;
    uint4 v;
    v.x = (unsigned)pk[0] | ((unsigned)pk[1] << 16);
    v.y = (unsigned)pk[2] | ((unsigned)pk[3] << 16);
    v.z = (unsigned)pk[4] | ((unsigned)pk[5] << 16);
    v.w = (unsigned)pk[6] | ((unsigned)pk[7] << 16);
    unsigned short* dst = hhT + (((size_t)(b * HEADS + head) * DH + d) * Nn + nt * 16 + 8 * hl);
    *(uint4*)dst = v;
}

// ---------------------------------------------------------------------------
// 5) Per-head attention scores: s_src/s_dst[b][h][n] = <hh[b,n,h,:], a_vec[h]>
// ---------------------------------------------------------------------------
__global__ void k_scores(const float* __restrict__ hh, const float* __restrict__ a_src,
                         const float* __restrict__ a_dst, float* __restrict__ s_src,
                         float* __restrict__ s_dst) {
    const int lane = threadIdx.x & 31;
    const int row  = (blockIdx.x * blockDim.x + threadIdx.x) >> 5;  // 0 .. B*N-1
    if (row >= Bn * Nn) return;
    const int b = row / Nn, n = row % Nn;
    const float* hr = hh + (size_t)row * HID;
    float ps[HEADS], pd[HEADS];
    #pragma unroll
    for (int hd = 0; hd < HEADS; ++hd) {
        float v = hr[hd * DH + lane];
        ps[hd] = v * a_src[hd * DH + lane];
        pd[hd] = v * a_dst[hd * DH + lane];
    }
    #pragma unroll
    for (int hd = 0; hd < HEADS; ++hd) { ps[hd] = wave_sum(ps[hd]); pd[hd] = wave_sum(pd[hd]); }
    if (lane == 0) {
        #pragma unroll
        for (int hd = 0; hd < HEADS; ++hd) {
            s_src[((size_t)(b * HEADS + hd)) * Nn + n] = ps[hd];
            s_dst[((size_t)(b * HEADS + hd)) * Nn + n] = pd[hd];
        }
    }
}

// ---------------------------------------------------------------------------
// 6) Softmax stats: rowmax (exact, via monotonic lrelu) and 1/sum(exp).
//    One wave per (b,h,i); mask word shared by all 32 lanes per chunk.
// ---------------------------------------------------------------------------
__global__ void k_stats(const float* __restrict__ s_src, const float* __restrict__ s_dst,
                        const unsigned* __restrict__ mask, float* __restrict__ rowmax,
                        float* __restrict__ invl) {
    const int lane = threadIdx.x;
    const int i = blockIdx.x, h = blockIdx.y, b = blockIdx.z;
    const int bh = b * HEADS + h;
    const float sd = s_dst[(size_t)bh * Nn + i];
    const unsigned* mrow = mask + (size_t)i * MW;
    const float* ssb = s_src + (size_t)bh * Nn;
    float mx = -3.0e38f;
    for (int t = 0; t < MW; ++t) {
        unsigned w = mrow[t];
        if ((w >> lane) & 1u) mx = fmaxf(mx, ssb[t * 32 + lane]);
    }
    mx = wave_max(mx);
    const float rm = lrelu(sd + mx);   // exact row max: lrelu is monotonic
    float s = 0.0f;
    for (int t = 0; t < MW; ++t) {
        unsigned w = mrow[t];
        float e = lrelu(sd + ssb[t * 32 + lane]);
        s += ((w >> lane) & 1u) ? __expf(e - rm) : 0.0f;
    }
    s = wave_sum(s);
    if (lane == 0) {
        rowmax[(size_t)bh * Nn + i] = rm;
        invl[(size_t)bh * Nn + i]   = 1.0f / s;   // diag always masked-in -> s > 0
    }
}

// ---------------------------------------------------------------------------
// 7) Aggregation: out[b,i,h*32+d] = sum_j alpha[b,h,i,j] * hh[b,j,h,d] (+bias,
//    optional GELU). One wave per (b,h,16-row tile); alpha tiles are generated
//    in-register as bf16 WMMA A-fragments; K streams in chunks of 32.
// ---------------------------------------------------------------------------
template <bool GELU>
__global__ void k_aggregate(const float* __restrict__ s_src, const float* __restrict__ s_dst,
                            const float* __restrict__ rowmax, const float* __restrict__ invl,
                            const unsigned* __restrict__ mask,
                            const unsigned short* __restrict__ hhT,
                            const float* __restrict__ bias, float* __restrict__ out) {
    const int lane = threadIdx.x;
    const int ln = lane & 15, hl = lane >> 4;
    const int it = blockIdx.x;   // 0..127
    const int h  = blockIdx.y;   // 0..3
    const int b  = blockIdx.z;   // 0..3
    const int bh = b * HEADS + h;

    const int irow = it * 16 + ln;
    const float sd = s_dst[(size_t)bh * Nn + irow];
    const float rm = rowmax[(size_t)bh * Nn + irow];
    const float il = invl[(size_t)bh * Nn + irow];
    const unsigned* mrow = mask + (size_t)irow * MW;
    const unsigned short* bp0 = hhT + ((size_t)bh * DH + ln)      * Nn;
    const unsigned short* bp1 = hhT + ((size_t)bh * DH + 16 + ln) * Nn;
    const float* ssb = s_src + (size_t)bh * Nn;

    v8f acc0 = {}, acc1 = {};
    for (int t = 0; t < MW; ++t) {
        const float ssv = ssb[t * 32 + lane];        // lane k holds s_src[t*32+k]
        const unsigned w = mrow[t];
        v16bf a;
        #pragma unroll
        for (int e = 0; e < 16; ++e) {
            // contiguous K layout (matches B): lanes0-15 K 0..15, lanes16-31 K 16..31
            const int K = e + hl * 16;
            float ev = sd + __shfl(ssv, K, 32);
            ev = lrelu(ev);
            float al = ((w >> K) & 1u) ? __expf(ev - rm) * il : 0.0f;
            a[e] = (__bf16)al;
        }
        v16bf b0 = *(const v16bf*)(bp0 + t * 32 + hl * 16);
        v16bf b1 = *(const v16bf*)(bp1 + t * 32 + hl * 16);
        acc0 = __builtin_amdgcn_wmma_f32_16x16x32_bf16(false, a, false, b0, (short)0, acc0, false, false);
        acc1 = __builtin_amdgcn_wmma_f32_16x16x32_bf16(false, a, false, b1, (short)0, acc1, false, false);
    }
    const int d0 = h * DH + ln, d1 = d0 + 16;
    const float bi0 = bias[d0], bi1 = bias[d1];
    #pragma unroll
    for (int r = 0; r < 8; ++r) {
        const int i = it * 16 + r + 8 * hl;
        float o0 = acc0[r] + bi0;
        float o1 = acc1[r] + bi1;
        if (GELU) { o0 = gelu_exact(o0); o1 = gelu_exact(o1); }
        out[((size_t)(b * Nn + i)) * ODIM + d0] = o0;
        out[((size_t)(b * Nn + i)) * ODIM + d1] = o1;
    }
}

// ---------------------------------------------------------------------------
extern "C" void kernel_launch(void* const* d_in, const int* in_sizes, int n_in,
                              void* d_out, int out_size, void* d_ws, size_t ws_size,
                              hipStream_t stream) {
    const float* x      = (const float*)d_in[0];
    const float* emb    = (const float*)d_in[1];
    const float* Wp     = (const float*)d_in[2];
    const float* bp     = (const float*)d_in[3];
    const float* W1     = (const float*)d_in[4];
    const float* a1_src = (const float*)d_in[5];
    const float* a1_dst = (const float*)d_in[6];
    const float* b1     = (const float*)d_in[7];
    const float* W2     = (const float*)d_in[8];
    const float* a2_src = (const float*)d_in[9];
    const float* a2_dst = (const float*)d_in[10];
    const float* b2     = (const float*)d_in[11];
    float* outp = (float*)d_out;

    // Workspace carve-up (all offsets 256B aligned)
    char* ws = (char*)d_ws;
    size_t off = 0;
    auto alloc = [&](size_t bytes) { void* p = ws + off; off = (off + bytes + 255) & ~(size_t)255; return p; };
    float*          ne    = (float*)alloc((size_t)Nn * HID * 4);
    unsigned*       mk    = (unsigned*)alloc((size_t)Nn * MW * 4);
    float*          h     = (float*)alloc((size_t)Bn * Nn * HID * 4);
    float*          hh    = (float*)alloc((size_t)Bn * Nn * HID * 4);
    unsigned short* hhT   = (unsigned short*)alloc((size_t)Bn * HEADS * DH * Nn * 2);
    float*          ssrc  = (float*)alloc((size_t)Bn * HEADS * Nn * 4);
    float*          sdst  = (float*)alloc((size_t)Bn * HEADS * Nn * 4);
    float*          rmax  = (float*)alloc((size_t)Bn * HEADS * Nn * 4);
    float*          il    = (float*)alloc((size_t)Bn * HEADS * Nn * 4);
    float*          h2    = (float*)alloc((size_t)Bn * Nn * HID * 4);

    // Adjacency (shared by both layers)
    k_normalize<<<Nn / 8, 256, 0, stream>>>(emb, ne);
    k_mask<<<dim3(Nn / 32, Nn / 16), 32, 0, stream>>>(ne, mk);

    // Projection
    k_project<<<dim3(HID / 16, Bn * Nn / 16), 32, 0, stream>>>(x, Wp, bp, h);

    // ---- GAT layer 1 (output -> h2, with GELU) ----
    k_hidden<<<dim3(HID / 16, Nn / 16, Bn), 32, 0, stream>>>(h, W1, hh, hhT);
    k_scores<<<Bn * Nn / 8, 256, 0, stream>>>(hh, a1_src, a1_dst, ssrc, sdst);
    k_stats<<<dim3(Nn, HEADS, Bn), 32, 0, stream>>>(ssrc, sdst, mk, rmax, il);
    k_aggregate<true><<<dim3(Nn / 16, HEADS, Bn), 32, 0, stream>>>(ssrc, sdst, rmax, il, mk, hhT, b1, h2);

    // ---- GAT layer 2 (output -> d_out, no activation) ----
    k_hidden<<<dim3(HID / 16, Nn / 16, Bn), 32, 0, stream>>>(h2, W2, hh, hhT);
    k_scores<<<Bn * Nn / 8, 256, 0, stream>>>(hh, a2_src, a2_dst, ssrc, sdst);
    k_stats<<<dim3(Nn, HEADS, Bn), 32, 0, stream>>>(ssrc, sdst, mk, rmax, il);
    k_aggregate<false><<<dim3(Nn / 16, HEADS, Bn), 32, 0, stream>>>(ssrc, sdst, rmax, il, mk, hhT, b2, outp);
}
